// NonLocalBlock_87548613362307
// MI455X (gfx1250) — matrix operand order repaired and checked by reading, not compiled, not run on hardware
//
#include <hip/hip_runtime.h>

typedef __attribute__((ext_vector_type(16))) __bf16 v16bf;
typedef __attribute__((ext_vector_type(8)))  __bf16 v8bf;
typedef __attribute__((ext_vector_type(8)))  float  v8f;
typedef __attribute__((ext_vector_type(4)))  unsigned u32x4;
typedef __attribute__((ext_vector_type(8)))  int      i32x8;
typedef __attribute__((ext_vector_type(4)))  int      i32x4;

#define NB 4
#define NC 256
#define NCI 128
#define NN 4096

__device__ __forceinline__ v8f vzero8() {
  v8f v = {0.f,0.f,0.f,0.f,0.f,0.f,0.f,0.f};
  return v;
}

__device__ __forceinline__ v8f wmma_bf16(v16bf a, v16bf b, v8f c) {
  return __builtin_amdgcn_wmma_f32_16x16x32_bf16(false, a, false, b, (short)0, c, false, false);
}

// ---------------- CDNA5 async global->LDS staging (ASYNCcnt) ----------------
__device__ __forceinline__ void async_ld16(void* lds, const void* gm) {
  unsigned loff = (unsigned)(size_t)lds;
  asm volatile("global_load_async_to_lds_b128 %0, %1, off"
               :: "v"(loff), "v"(gm) : "memory");
}
__device__ __forceinline__ void async_wait0() {
  asm volatile("s_wait_asynccnt 0x0" ::: "memory");
}
__device__ __forceinline__ void async_wait4() { // oldest 4 done (in-order)
  asm volatile("s_wait_asynccnt 0x4" ::: "memory");
}

// ------------- Tensor Data Mover: 2D tile load (D# per ISA ch.8) -------------
// Loads tile_y rows of tile_x bf16 elements from a row-major tensor
// (row stride stride_x elements) into contiguous LDS at lds_off.
// amdgpu-toolchain (clang-23) 6-arg builtin form.
__device__ __forceinline__ void tdm_load_2d_bf16(unsigned lds_off,
                                                 const void* gaddr,
                                                 unsigned tile_x, unsigned tile_y,
                                                 unsigned tensor_x,
                                                 unsigned tensor_y,
                                                 unsigned stride_x) {
  const unsigned long long ga = (unsigned long long)(size_t)gaddr;
  u32x4 g0;
  g0[0] = 1u;                                            // count=1, user D#
  g0[1] = lds_off;                                       // lds_addr (bytes)
  g0[2] = (unsigned)(ga & 0xFFFFFFFFu);                  // global_addr[31:0]
  g0[3] = (unsigned)((ga >> 32) & 0x01FFFFFFu) | (2u << 30); // addr hi | type=2
  i32x8 g1;
  g1[0] = (int)(1u << 16);                               // data_size=1 (2 bytes)
  g1[1] = (int)((tensor_x & 0xFFFFu) << 16);             // tensor_dim0 lo
  g1[2] = (int)((tensor_x >> 16) | ((tensor_y & 0xFFFFu) << 16)); // dim0 hi|dim1 lo
  g1[3] = (int)((tensor_y >> 16) | (tile_x << 16));      // dim1 hi | tile_dim0
  g1[4] = (int)tile_y;                                   // tile_dim1 (tile_dim2=0)
  g1[5] = (int)stride_x;                                 // tensor_dim0_stride lo
  g1[6] = 0;                                             // stride0 hi | stride1 lo
  g1[7] = 0;
  const i32x4 z4 = {0, 0, 0, 0};                         // groups 2/3 unused
  const i32x8 z8 = {0, 0, 0, 0, 0, 0, 0, 0};
  __builtin_amdgcn_tensor_load_to_lds(g0, g1, z4, z4, z8, 0);
}

// ---- WMMA fragment loaders: table k order per lane (ds_load_b128 at k0+8h and k0+16+8h) ----
__device__ __forceinline__ v16bf frag_a_row(const __bf16* src, int m0, int k0,
                                            int lda, int lane) {
  const int m  = m0 + (lane & 15);
  const int kh = k0 + ((lane >> 4) << 3);
  const __bf16* p = src + m * lda + kh;
  v8bf lo = *(const v8bf*)p;
  v8bf hi = *(const v8bf*)(p + 16);
  return __builtin_shufflevector(lo, hi, 0,1,2,3,4,5,6,7,8,9,10,11,12,13,14,15);
}
__device__ __forceinline__ v16bf frag_b_nmaj(const __bf16* src, int k0, int n0,
                                             int ldn, int lane) {
  const int n  = n0 + (lane & 15);
  const int kb = k0 + ((lane >> 4) << 3);
  const __bf16* p = src + n * ldn + kb;
  v8bf lo = *(const v8bf*)p;
  v8bf hi = *(const v8bf*)(p + 16);
  return __builtin_shufflevector(lo, hi, 0,1,2,3,4,5,6,7,8,9,10,11,12,13,14,15);
}

// ------------------------- K0a: f32 -> bf16 convert -------------------------
__global__ __launch_bounds__(256) void cvt_bf16(const float* __restrict__ src,
                                                __bf16* __restrict__ dst, int n) {
  int i = blockIdx.x * 256 + threadIdx.x;
  if (i < n) dst[i] = (__bf16)src[i];
}

// ---------------- K0b: x [B][C][N] f32 -> xT [B][N][C] bf16 ----------------
__global__ __launch_bounds__(256) void transpose_x(const float* __restrict__ x,
                                                   __bf16* __restrict__ xT) {
  const int bz = blockIdx.z;
  const int c0 = blockIdx.y * 32;
  const int n0 = blockIdx.x * 32;
  __shared__ float t[32][33];
  const int tx = threadIdx.x & 31, ty = threadIdx.x >> 5;
#pragma unroll
  for (int i = 0; i < 4; ++i)
    t[ty + i * 8][tx] = x[((size_t)bz * NC + c0 + ty + i * 8) * NN + n0 + tx];
  __syncthreads();
#pragma unroll
  for (int i = 0; i < 4; ++i)
    xT[((size_t)bz * NN + n0 + ty + i * 8) * NC + c0 + tx] =
        (__bf16)t[tx][ty + i * 8];
}

// ------------------- K1: QKV projection GEMM (bf16 WMMA) -------------------
// Double-buffered async staging; theta/phi -> [n][ci], g -> [ci][n].
__global__ __launch_bounds__(256) void qkv_gemm(
    const __bf16* __restrict__ w_theta, const __bf16* __restrict__ w_phi,
    const __bf16* __restrict__ w_g, const float* __restrict__ b_theta,
    const float* __restrict__ b_phi, const float* __restrict__ b_g,
    const __bf16* __restrict__ xT, __bf16* __restrict__ qkv) {
  const int n0   = blockIdx.x * 128;
  const int s    = blockIdx.y;
  const int bz   = blockIdx.z;
  const int tid  = threadIdx.x;
  const int lane = tid & 31;
  const int wave = tid >> 5;
  const int wm   = wave & 1;
  const int wn   = wave >> 1;
  const int lq   = lane & 15;
  const int lh   = (lane >> 4) << 3;

  const __bf16* w    = (s == 0) ? w_theta : (s == 1) ? w_phi : w_g;
  const float*  bias = (s == 0) ? b_theta : (s == 1) ? b_phi : b_g;
  const __bf16* xbat = xT + (size_t)bz * NN * NC;

  __shared__ __align__(32) __bf16 As[2][128 * 32];
  __shared__ __align__(32) __bf16 Bs[2][128 * 32];

  v8f acc[4][2];
#pragma unroll
  for (int i = 0; i < 4; ++i)
#pragma unroll
    for (int j = 0; j < 2; ++j) acc[i][j] = vzero8();

  const int row = tid >> 1, seg = tid & 1;
  auto stage = [&](int kt, int buf) {
    const int kk = kt * 32;
    async_ld16(&As[buf][row * 32 + seg * 16], &w[row * 256 + kk + seg * 16]);
    async_ld16(&As[buf][row * 32 + seg * 16 + 8],
               &w[row * 256 + kk + seg * 16 + 8]);
    async_ld16(&Bs[buf][row * 32 + seg * 16],
               &xbat[(size_t)(n0 + row) * NC + kk + seg * 16]);
    async_ld16(&Bs[buf][row * 32 + seg * 16 + 8],
               &xbat[(size_t)(n0 + row) * NC + kk + seg * 16 + 8]);
  };

  stage(0, 0);
  for (int kt = 0; kt < 8; ++kt) {
    const int cur = kt & 1;
    if (kt < 7) { stage(kt + 1, cur ^ 1); async_wait4(); }
    else        { async_wait0(); }
    __syncthreads();

    const __bf16* Amat = (s < 2) ? As[cur] : Bs[cur]; // swap roles for g
    const __bf16* Bmat = (s < 2) ? Bs[cur] : As[cur];
    v16bf afr[4], bfr[2];
#pragma unroll
    for (int mt = 0; mt < 4; ++mt)
      afr[mt] = frag_a_row(Amat, wm * 64 + mt * 16, 0, 32, lane);
#pragma unroll
    for (int nt = 0; nt < 2; ++nt)
      bfr[nt] = frag_b_nmaj(Bmat, 0, wn * 32 + nt * 16, 32, lane);
#pragma unroll
    for (int mt = 0; mt < 4; ++mt)
#pragma unroll
      for (int nt = 0; nt < 2; ++nt)
        acc[mt][nt] = wmma_bf16(afr[mt], bfr[nt], acc[mt][nt]);
    __syncthreads();
  }

  const size_t obase = (size_t)(bz * 3 + s) * NCI * NN;
  if (s < 2) {
#pragma unroll
    for (int mt = 0; mt < 4; ++mt) {
      const int mbase = wm * 64 + mt * 16 + lh;
      const float4 bA = *(const float4*)&bias[mbase];
      const float4 bB = *(const float4*)&bias[mbase + 4];
      const float bb[8] = {bA.x, bA.y, bA.z, bA.w, bB.x, bB.y, bB.z, bB.w};
#pragma unroll
      for (int nt = 0; nt < 2; ++nt) {
        const int n = n0 + wn * 32 + nt * 16 + lq;
        v8bf o;
#pragma unroll
        for (int v = 0; v < 8; ++v) o[v] = (__bf16)(acc[mt][nt][v] + bb[v]);
        *(v8bf*)&qkv[obase + (size_t)n * NCI + mbase] = o;
      }
    }
  } else {
#pragma unroll
    for (int nt = 0; nt < 2; ++nt) {
      const int o  = wn * 32 + nt * 16 + lq;
      const float bo = bias[o];
#pragma unroll
      for (int mt = 0; mt < 4; ++mt) {
        const int nbase = n0 + wm * 64 + mt * 16 + lh;
        v8bf ov;
#pragma unroll
        for (int v = 0; v < 8; ++v) ov[v] = (__bf16)(acc[mt][nt][v] + bo);
        *(v8bf*)&qkv[obase + (size_t)o * NN + nbase] = ov;
      }
    }
  }
}

// -------- K2: fused flash attention: softmax(theta^T phi) @ g^T -> yT --------
// phi chunk: async-to-LDS; g chunk: Tensor Data Mover 2D descriptor,
// overlapped with logits+softmax on TENSORcnt.
__global__ __launch_bounds__(256) void attention(const __bf16* __restrict__ qkv,
                                                 __bf16* __restrict__ yT) {
  const int b    = blockIdx.y;
  const int q0   = blockIdx.x * 64;
  const int tid  = threadIdx.x;
  const int lane = tid & 31;
  const int wave = tid >> 5;
  const int qt   = wave >> 1;
  const int kt2  = wave & 1;
  const int cb   = (wave & 1) * 64;
  const int lq   = lane & 15;
  const int lh   = (lane >> 4) << 3;
  const int q_loc = qt * 16 + lq;

  __shared__ __align__(32) __bf16 Qs[64 * 128];  // theta^T [q][ci]
  __shared__ __align__(32) __bf16 Ks[64 * 128];  // phi^T  [key][ci]
  __shared__ __align__(32) __bf16 Gs[128 * 64];  // g      [ci][key] (TDM dest)
  __shared__ __align__(32) __bf16 Ps[64 * 64];   // probs  [q][key]
  __shared__ float pmax[8][16], psum[8][16];
  __shared__ float rsum[64], rmax[64], rscale[64];

  const __bf16* thetaT = qkv + (size_t)(b * 3 + 0) * NCI * NN;
  const __bf16* phiT   = qkv + (size_t)(b * 3 + 1) * NCI * NN;
  const __bf16* g      = qkv + (size_t)(b * 3 + 2) * NCI * NN;

  for (int idx = tid; idx < 64 * 16; idx += 256) {
    const int q = idx >> 4, seg = idx & 15;
    async_ld16(&Qs[q * 128 + seg * 8],
               &thetaT[(size_t)(q0 + q) * NCI + seg * 8]);
  }
  if (tid < 64) { rsum[tid] = 0.f; rmax[tid] = -3.0e38f; }

  v8f yacc[4];
#pragma unroll
  for (int t = 0; t < 4; ++t) yacc[t] = vzero8();

  for (int m0 = 0; m0 < NN; m0 += 64) {
    // ---- issue phi chunk (async) + g chunk (TDM), overlap with compute ----
    for (int idx = tid; idx < 64 * 16; idx += 256) {
      const int key = idx >> 4, seg = idx & 15;
      async_ld16(&Ks[key * 128 + seg * 8],
                 &phiT[(size_t)(m0 + key) * NCI + seg * 8]);
    }
    if (wave == 0)
      tdm_load_2d_bf16((unsigned)(size_t)&Gs[0], g + m0,
                       /*tile_x=*/64, /*tile_y=*/NCI,
                       /*tensor_x=*/NN, /*tensor_y=*/NCI, /*stride_x=*/NN);
    if (m0 + 64 < NN) // warm L2 for next chunk (global_prefetch_b8)
      __builtin_prefetch(&phiT[(size_t)(m0 + 64 + (tid & 63)) * NCI], 0, 1);
    async_wait0();          // Qs (first iter) + phi ready; g still in flight
    __syncthreads();

    // ---- logits: D=[key][q] ----
    v8f l0 = vzero8(), l1 = vzero8();
#pragma unroll
    for (int kt = 0; kt < 4; ++kt) {
      const int kk = kt * 32;
      v16bf bq = frag_b_nmaj(Qs, kk, qt * 16, 128, lane);
      v16bf a0 = frag_a_row(Ks, kt2 * 32, kk, 128, lane);
      v16bf a1 = frag_a_row(Ks, kt2 * 32 + 16, kk, 128, lane);
      l0 = wmma_bf16(a0, bq, l0);
      l1 = wmma_bf16(a1, bq, l1);
    }

    // ---- chunk max via lane reduction + shfl ----
    float mx = -3.0e38f;
#pragma unroll
    for (int v = 0; v < 8; ++v) mx = fmaxf(mx, fmaxf(l0[v], l1[v]));
    mx = fmaxf(mx, __shfl_xor(mx, 16, 32));
    if (lane < 16) pmax[wave][lq] = mx;
    __syncthreads();

    const float cmax = fmaxf(pmax[2 * qt][lq], pmax[2 * qt + 1][lq]);
    const float mnew = fmaxf(rmax[q_loc], cmax);
    v8bf p0, p1;
    float ssum = 0.f;
#pragma unroll
    for (int v = 0; v < 8; ++v) {
      const float e0 = __expf(l0[v] - mnew);
      const float e1 = __expf(l1[v] - mnew);
      p0[v] = (__bf16)e0;
      p1[v] = (__bf16)e1;
      ssum += e0 + e1;
    }
    *(v8bf*)&Ps[q_loc * 64 + kt2 * 32 + lh]      = p0;
    *(v8bf*)&Ps[q_loc * 64 + kt2 * 32 + 16 + lh] = p1;
    ssum += __shfl_xor(ssum, 16, 32);
    if (lane < 16) psum[wave][lq] = ssum;
    __syncthreads();

    if (tid < 64) { // running stats update
      const int mq = tid >> 4, l = tid & 15;
      const float cm = fmaxf(pmax[2 * mq][l], pmax[2 * mq + 1][l]);
      const float mo = rmax[tid];
      const float mn = fmaxf(mo, cm);
      const float sc = __expf(mo - mn);
      rsum[tid]   = rsum[tid] * sc + psum[2 * mq][l] + psum[2 * mq + 1][l];
      rmax[tid]   = mn;
      rscale[tid] = sc;
    }
    if (wave == 0) __builtin_amdgcn_s_wait_tensorcnt(0); // drain TDM (g ready)
    __syncthreads();

    const float sc = rscale[q_loc];
#pragma unroll
    for (int t = 0; t < 4; ++t)
#pragma unroll
      for (int v = 0; v < 8; ++v) yacc[t][v] *= sc;

    // ---- PV: D=[ci][q]: A=g rows, B=P ----
#pragma unroll
    for (int kt = 0; kt < 2; ++kt) {
      const int kk = kt * 32;
      v16bf bp = frag_b_nmaj(Ps, kk, qt * 16, 64, lane);
#pragma unroll
      for (int t = 0; t < 4; ++t) {
        v16bf ag = frag_a_row(Gs, cb + t * 16, kk, 64, lane);
        yacc[t] = wmma_bf16(ag, bp, yacc[t]);
      }
    }
    __syncthreads();
  }

  const float inv = 1.0f / rsum[q_loc];
#pragma unroll
  for (int t = 0; t < 4; ++t) {
    v8bf o;
#pragma unroll
    for (int v = 0; v < 8; ++v) o[v] = (__bf16)(yacc[t][v] * inv);
    *(v8bf*)&yT[((size_t)b * NN + q0 + q_loc) * NCI + cb + t * 16 + lh] = o;
  }
}

// ------------- K3: output projection Wy = W_w @ y + W_b (f32 out) -------------
__global__ __launch_bounds__(256) void out_gemm(const __bf16* __restrict__ wW,
                                                const float* __restrict__ Wb,
                                                const __bf16* __restrict__ yT,
                                                float* __restrict__ Wy) {
  const int n0    = blockIdx.x * 128;
  const int mbase = blockIdx.y * 128;
  const int bz    = blockIdx.z;
  const int tid   = threadIdx.x;
  const int lane  = tid & 31;
  const int wave  = tid >> 5;
  const int wm    = wave & 1;
  const int wn    = wave >> 1;
  const int lq    = lane & 15;
  const int lh    = (lane >> 4) << 3;

  const __bf16* ybat = yT + (size_t)bz * NN * NCI;

  __shared__ __align__(32) __bf16 As[2][128 * 32]; // W rows [c][k]
  __shared__ __align__(32) __bf16 Bs[2][128 * 32]; // y rows [n][k]

  v8f acc[4][2];
#pragma unroll
  for (int i = 0; i < 4; ++i)
#pragma unroll
    for (int j = 0; j < 2; ++j) acc[i][j] = vzero8();

  const int row = tid >> 1, seg = tid & 1;
  auto stage = [&](int kt, int buf) {
    const int kk = kt * 32;
    async_ld16(&As[buf][row * 32 + seg * 16],
               &wW[(size_t)(mbase + row) * NCI + kk + seg * 16]);
    async_ld16(&As[buf][row * 32 + seg * 16 + 8],
               &wW[(size_t)(mbase + row) * NCI + kk + seg * 16 + 8]);
    async_ld16(&Bs[buf][row * 32 + seg * 16],
               &ybat[(size_t)(n0 + row) * NCI + kk + seg * 16]);
    async_ld16(&Bs[buf][row * 32 + seg * 16 + 8],
               &ybat[(size_t)(n0 + row) * NCI + kk + seg * 16 + 8]);
  };

  stage(0, 0);
  for (int kt = 0; kt < 4; ++kt) {
    const int cur = kt & 1;
    if (kt < 3) { stage(kt + 1, cur ^ 1); async_wait4(); }
    else        { async_wait0(); }
    __syncthreads();

    v16bf afr[4], bfr[2];
#pragma unroll
    for (int mtt = 0; mtt < 4; ++mtt)
      afr[mtt] = frag_a_row(Bs[cur], wm * 64 + mtt * 16, 0, 32, lane);
#pragma unroll
    for (int nt = 0; nt < 2; ++nt)
      bfr[nt] = frag_b_nmaj(As[cur], 0, wn * 32 + nt * 16, 32, lane);
#pragma unroll
    for (int mtt = 0; mtt < 4; ++mtt)
#pragma unroll
      for (int nt = 0; nt < 2; ++nt)
        acc[mtt][nt] = wmma_bf16(afr[mtt], bfr[nt], acc[mtt][nt]);
    __syncthreads();
  }

#pragma unroll
  for (int nt = 0; nt < 2; ++nt) {
    const int c  = mbase + wn * 32 + nt * 16 + lq;
    const float wb = Wb[c];
#pragma unroll
    for (int mtt = 0; mtt < 4; ++mtt) {
      const int nbase = n0 + wm * 64 + mtt * 16 + lh;
      float* dst = &Wy[((size_t)bz * NC + c) * NN + nbase];
      const v8f a = acc[mtt][nt];
      *(float4*)dst       = make_float4(a[0] + wb, a[1] + wb, a[2] + wb, a[3] + wb);
      *(float4*)(dst + 4) = make_float4(a[4] + wb, a[5] + wb, a[6] + wb, a[7] + wb);
    }
  }
}

// ---------------- K4: per-channel BatchNorm statistics ----------------
__global__ __launch_bounds__(256) void bn_stats(const float* __restrict__ Wy,
                                                float* __restrict__ stats) {
  const int c   = blockIdx.x;
  const int tid = threadIdx.x;
  float s = 0.f, s2 = 0.f;
  for (int i = tid; i < NB * NN; i += 256) {
    const int b = i >> 12, n = i & (NN - 1);
    const float v = Wy[((size_t)b * NC + c) * NN + n];
    s += v; s2 += v * v;
  }
  __shared__ float sh[256], sh2[256];
  sh[tid] = s; sh2[tid] = s2;
  __syncthreads();
  for (int off = 128; off > 0; off >>= 1) {
    if (tid < off) { sh[tid] += sh[tid + off]; sh2[tid] += sh2[tid + off]; }
    __syncthreads();
  }
  if (tid == 0) {
    const float inv = 1.0f / (float)(NB * NN);
    const float mean = sh[0] * inv;
    stats[c]      = mean;
    stats[NC + c] = sh2[0] * inv - mean * mean;
  }
}

// ---------------- K5: BN apply + residual ----------------
__global__ __launch_bounds__(256) void bn_apply(const float* __restrict__ Wy,
                                                const float* __restrict__ x,
                                                const float* __restrict__ stats,
                                                const float* __restrict__ gamma,
                                                const float* __restrict__ beta,
                                                float* __restrict__ out) {
  const size_t i = (size_t)blockIdx.x * 256 + threadIdx.x;
  const int c = (int)((i >> 12) & (NC - 1));
  const float mean = stats[c], var = stats[NC + c];
  out[i] = gamma[c] * (Wy[i] - mean) * rsqrtf(var + 1e-5f) + beta[c] + x[i];
}

extern "C" void kernel_launch(void* const* d_in, const int* in_sizes, int n_in,
                              void* d_out, int out_size, void* d_ws, size_t ws_size,
                              hipStream_t stream) {
  (void)in_sizes; (void)n_in; (void)out_size; (void)ws_size;
  const float* x    = (const float*)d_in[0];
  const float* g_w  = (const float*)d_in[1];
  const float* g_b  = (const float*)d_in[2];
  const float* th_w = (const float*)d_in[3];
  const float* th_b = (const float*)d_in[4];
  const float* ph_w = (const float*)d_in[5];
  const float* ph_b = (const float*)d_in[6];
  const float* W_w  = (const float*)d_in[7];
  const float* W_b  = (const float*)d_in[8];
  const float* gam  = (const float*)d_in[9];
  const float* bet  = (const float*)d_in[10];
  float* out = (float*)d_out;

  char* ws = (char*)d_ws;
  const size_t sz_xT  = (size_t)NB * NN * NC * 2;
  const size_t sz_w   = (size_t)NCI * NC * 2;
  const size_t sz_qkv = (size_t)NB * 3 * NCI * NN * 2;
  const size_t sz_y   = (size_t)NB * NN * NCI * 2;
  const size_t sz_Wy  = (size_t)NB * NC * NN * 4;

  __bf16* xT  = (__bf16*)(ws);
  __bf16* wt  = (__bf16*)(ws + sz_xT);
  __bf16* wp  = (__bf16*)(ws + sz_xT + sz_w);
  __bf16* wg  = (__bf16*)(ws + sz_xT + 2 * sz_w);
  __bf16* wWb = (__bf16*)(ws + sz_xT + 3 * sz_w);
  __bf16* qkv = (__bf16*)(ws + sz_xT + 4 * sz_w);
  __bf16* yTb = (__bf16*)(ws + sz_xT + 4 * sz_w + sz_qkv);
  float*  Wy  = (float*) (ws + sz_xT + 4 * sz_w + sz_qkv + sz_y);
  float*  st  = (float*) (ws + sz_xT + 4 * sz_w + sz_qkv + sz_y + sz_Wy);

  transpose_x<<<dim3(NN / 32, NC / 32, NB), 256, 0, stream>>>(x, xT);
  cvt_bf16<<<128, 256, 0, stream>>>(th_w, wt, NCI * NC);
  cvt_bf16<<<128, 256, 0, stream>>>(ph_w, wp, NCI * NC);
  cvt_bf16<<<128, 256, 0, stream>>>(g_w,  wg, NCI * NC);
  cvt_bf16<<<128, 256, 0, stream>>>(W_w,  wWb, NC * NCI);

  qkv_gemm<<<dim3(NN / 128, 3, NB), 256, 0, stream>>>(wt, wp, wg, th_b, ph_b,
                                                      g_b, xT, qkv);
  attention<<<dim3(NN / 64, NB), 256, 0, stream>>>(qkv, yTb);
  out_gemm<<<dim3(NN / 128, 2, NB), 256, 0, stream>>>(wWb, W_b, yTb, Wy);
  bn_stats<<<NC, 256, 0, stream>>>(Wy, st);

  const int nx = NB * NC * NN;
  bn_apply<<<nx / 256, 256, 0, stream>>>(Wy, x, st, gam, bet, out);
}
